// Encoder_VGAE_21045339751000
// MI455X (gfx1250) — compile-verified, matrix operand order checked by reading, not run on hardware
//
#include <hip/hip_runtime.h>
#include <hip/hip_bf16.h>

// ---------------------------------------------------------------------------
// VGAE encoder for MI455X (gfx1250, wave32, WMMA).
//  Phase A (bandwidth): edge scatter with global_atomic_add_f32.
//  Phase B (compute):   bf16 WMMA GEMMs (v_wmma_f32_16x16x32_bf16) with the
//                       16-row A tile staged in LDS via gfx1250 async-to-LDS
//                       (ASYNCcnt path), fused bias/ReLU/BatchNorm epilogues.
// ---------------------------------------------------------------------------

#define N_NODES   20000
#define IN_CH     512
#define HID       512
#define CAT       2560     // 5 * HID
#define OUT_CH    256
#define K1        1024     // mean(512) ++ x(512)
#define K2        5120     // mean2(2560) ++ h(2560)
#define BN_EPS    1e-5f

#define CK        1024     // K-chunk staged in LDS per barrier
#define SA_STRIDE 1032     // 1024 + 8 elem pad -> 2064B row stride (bank spread)

typedef __attribute__((ext_vector_type(16))) __bf16 v16bf;
typedef __attribute__((ext_vector_type(8)))  __bf16 v8bf;
typedef __attribute__((ext_vector_type(8)))  float  v8f;

// Exact types for the gfx1250 async-to-LDS builtin (from compiler diagnostic:
// param0 = "__vector_size__(4*sizeof(int)) int __device__ *").
typedef int v4i_gcc __attribute__((__vector_size__(16)));
typedef __attribute__((address_space(1))) v4i_gcc* gp_v4i;   // global (as1)
typedef __attribute__((address_space(3))) v4i_gcc* lp_v4i;   // LDS (as3)

#if __has_builtin(__builtin_amdgcn_global_load_async_to_lds_b128)
#define HAVE_ASYNC_LDS 1
#else
#define HAVE_ASYNC_LDS 0
#endif

// ---------------------------------------------------------------------------
// Fragment loader for 16-bit WMMA operands (wave32).
// A (16x32): lane L(0..15) = row L, holds K = {0..7, 16..23};
//            lane L(16..31) = row L-16, holds K = {8..15, 24..31}.
// B (32x16) mirrors this with lane -> column; since B = W^T, loading W's row
// (contiguous in K) with the same per-lane K pattern produces the B fragment.
// Works on global or LDS pointers (compiler infers ds_load_b128 for shared).
// ---------------------------------------------------------------------------
__device__ __forceinline__ v16bf load_frag(const __bf16* __restrict__ p, int lane) {
  const int base = (lane & 16) ? 8 : 0;
  v8bf lo = *(const v8bf*)(p + base);        // K = base .. base+7
  v8bf hi = *(const v8bf*)(p + base + 16);   // K = base+16 .. base+23
  v16bf f;
#pragma unroll
  for (int t = 0; t < 8; ++t) { f[t] = lo[t]; f[8 + t] = hi[t]; }
  return f;
}

// Cooperative stage of 64 bf16 (128B) from global `src` to LDS `dst`.
// Address-space conversion goes through integers: generic->as(1) keeps the
// 64-bit address; generic->as(3) truncates to the 32-bit LDS byte offset
// (ISA: LDS aperture address = addr[31:0]).
__device__ __forceinline__ void stage64(const __bf16* __restrict__ src,
                                        __bf16* __restrict__ dst) {
#if HAVE_ASYNC_LDS
#pragma unroll
  for (int t = 0; t < 8; ++t) {
    gp_v4i gp = (gp_v4i)(unsigned long long)(src + t * 8);
    lp_v4i lp = (lp_v4i)(unsigned int)(unsigned long long)(dst + t * 8);
    __builtin_amdgcn_global_load_async_to_lds_b128(gp, lp, 0, 0);
  }
#else
#pragma unroll
  for (int t = 0; t < 8; ++t)
    *(v8bf*)(dst + t * 8) = *(const v8bf*)(src + t * 8);
#endif
}

__device__ __forceinline__ void stage_wait() {
#if HAVE_ASYNC_LDS
#if __has_builtin(__builtin_amdgcn_s_wait_asynccnt)
  __builtin_amdgcn_s_wait_asynccnt(0);
#else
  asm volatile("s_wait_asynccnt 0x0" ::: "memory");
#endif
#endif
  __syncthreads();
}

// ---------------------------------------------------------------------------
// Elementwise / conversion kernels
// ---------------------------------------------------------------------------
__global__ void convert_x_kernel(const float* __restrict__ x,
                                 __bf16* __restrict__ xb, int total) {
  int i = blockIdx.x * blockDim.x + threadIdx.x;
  if (i < total) xb[i] = (__bf16)x[i];
}

// wcat1[k][j][c] : c<512 -> Wl5[k][j][c], else Wr5[k][j][c-512]
__global__ void build_wcat1_kernel(const float* __restrict__ Wl5,
                                   const float* __restrict__ Wr5,
                                   __bf16* __restrict__ wcat1) {
  int i = blockIdx.x * blockDim.x + threadIdx.x;
  if (i >= 5 * HID * K1) return;
  int c = i % K1;
  int j = (i / K1) % HID;
  int k = i / (K1 * HID);
  float v = (c < IN_CH) ? Wl5[((size_t)k * HID + j) * IN_CH + c]
                        : Wr5[((size_t)k * HID + j) * IN_CH + (c - IN_CH)];
  wcat1[i] = (__bf16)v;
}

// wcat2[j][p]: j<256 -> (p<2560 ? Wmu_l[j][p] : Wmu_r[j][p-2560])
//             j>=256 -> same with Wlv_*
__global__ void build_wcat2_kernel(const float* __restrict__ Wmu_l,
                                   const float* __restrict__ Wmu_r,
                                   const float* __restrict__ Wlv_l,
                                   const float* __restrict__ Wlv_r,
                                   __bf16* __restrict__ wcat2) {
  int i = blockIdx.x * blockDim.x + threadIdx.x;
  if (i >= 2 * OUT_CH * K2) return;
  int p = i % K2;
  int j = i / K2;
  const float* L = (j < OUT_CH) ? Wmu_l : Wlv_l;
  const float* R = (j < OUT_CH) ? Wmu_r : Wlv_r;
  int jj = j & (OUT_CH - 1);
  float v = (p < CAT) ? L[(size_t)jj * CAT + p] : R[(size_t)jj * CAT + (p - CAT)];
  wcat2[i] = (__bf16)v;
}

// ---------------------------------------------------------------------------
// Edge scatter kernels (segment sums via f32 atomics)
// ---------------------------------------------------------------------------
__global__ void scatter1_kernel(const float* __restrict__ x,
                                const int* __restrict__ src,
                                const int* __restrict__ dst,
                                const int* __restrict__ eattr,
                                float* __restrict__ acc,   // [N][2560]
                                float* __restrict__ cnt,   // [N][5]
                                int E) {
  int e = blockIdx.x;
  if (e >= E) return;
  int s = src[e], d = dst[e], k = eattr[e];
  const float* xs = x + (size_t)s * IN_CH;
  float* ad = acc + (size_t)d * CAT + (size_t)k * HID;
  for (int c = threadIdx.x; c < IN_CH; c += blockDim.x)
    atomicAdd(&ad[c], xs[c]);
  if (threadIdx.x == 0) atomicAdd(&cnt[d * 5 + k], 1.0f);
}

__global__ void scatter2_kernel(const __bf16* __restrict__ hb,
                                const int* __restrict__ src,
                                const int* __restrict__ dst,
                                float* __restrict__ acc,   // [N][2560]
                                float* __restrict__ cnt2,  // [N]
                                int E) {
  int e = blockIdx.x;
  if (e >= E) return;
  int s = src[e], d = dst[e];
  const __bf16* hs = hb + (size_t)s * CAT;
  float* ad = acc + (size_t)d * CAT;
  for (int c = threadIdx.x; c < CAT; c += blockDim.x)
    atomicAdd(&ad[c], (float)hs[c]);
  if (threadIdx.x == 0) atomicAdd(&cnt2[d], 1.0f);
}

// mean1: meanb[n][k*512+c] = acc[n][k*512+c] / max(cnt[n][k],1)
__global__ void mean1_kernel(const float* __restrict__ acc,
                             const float* __restrict__ cnt,
                             __bf16* __restrict__ meanb, int total) {
  int i = blockIdx.x * blockDim.x + threadIdx.x;
  if (i >= total) return;
  int ch = i % CAT;
  int n  = i / CAT;
  float c = cnt[n * 5 + (ch / HID)];
  meanb[i] = (__bf16)(acc[i] / fmaxf(c, 1.0f));
}

// mean2: meanb[n][c] = acc[n][c] / max(cnt2[n],1)
__global__ void mean2_kernel(const float* __restrict__ acc,
                             const float* __restrict__ cnt2,
                             __bf16* __restrict__ meanb, int total) {
  int i = blockIdx.x * blockDim.x + threadIdx.x;
  if (i >= total) return;
  int n = i / CAT;
  meanb[i] = (__bf16)(acc[i] / fmaxf(cnt2[n], 1.0f));
}

// ---------------------------------------------------------------------------
// GEMM1: per relation k, out[n][j] = relu( [mean_k | x](n,:) @ wcat1_k(j,:)^T
//        + bl5[k][j] ), then eval-BatchNorm, stored bf16 into h.
// grid = (N/16, 5), block = 256 (8 waves). Wave w -> cols [64w, 64w+64).
// A tile (16 x 1024) staged once in LDS, shared by all 8 waves.
// ---------------------------------------------------------------------------
__global__ void __launch_bounds__(256)
gemm1_kernel(const __bf16* __restrict__ meanb,   // [N][2560]
             const __bf16* __restrict__ xb,      // [N][512]
             const __bf16* __restrict__ wcat1,   // [5][512][1024]
             const float* __restrict__ bl5,      // [5][512]
             const float* __restrict__ gamma,
             const float* __restrict__ beta,
             const float* __restrict__ run_mean,
             const float* __restrict__ run_var,
             __bf16* __restrict__ hb) {          // [N][2560]
  __shared__ __bf16 sA[16 * SA_STRIDE];          // ~33 KB

  const int rowTile = blockIdx.x;
  const int k       = blockIdx.y;
  const int tid     = threadIdx.x;
  const int wave    = tid >> 5;
  const int lane    = tid & 31;
  const int row0    = rowTile * 16;
  const int j0      = wave * 64;

  // ---- stage A tile: thread t -> row t/16, 64-elem segment (t%16)*64 ----
  {
    const int r  = tid >> 4;            // 0..15
    const int c0 = (tid & 15) * 64;     // 0..960 (64-aligned; 512 boundary ok)
    const int n  = row0 + r;
    const __bf16* src = (c0 < IN_CH)
        ? (meanb + (size_t)n * CAT + (size_t)k * HID + c0)
        : (xb + (size_t)n * IN_CH + (c0 - IN_CH));
    stage64(src, sA + r * SA_STRIDE + c0);
  }
  stage_wait();

  const __bf16* aRowLds = sA + (lane & 15) * SA_STRIDE;
  const __bf16* wk = wcat1 + (size_t)k * HID * K1;
  const __bf16* bPtr[4];
#pragma unroll
  for (int t = 0; t < 4; ++t)
    bPtr[t] = wk + (size_t)(j0 + t * 16 + (lane & 15)) * K1;

  v8f acc0 = {}, acc1 = {}, acc2 = {}, acc3 = {};

  for (int kt = 0; kt < K1 / 32; ++kt) {
    v16bf a = load_frag(aRowLds + kt * 32, lane);     // ds_load path
    v16bf b0 = load_frag(bPtr[0] + kt * 32, lane);
    v16bf b1 = load_frag(bPtr[1] + kt * 32, lane);
    v16bf b2 = load_frag(bPtr[2] + kt * 32, lane);
    v16bf b3 = load_frag(bPtr[3] + kt * 32, lane);
    acc0 = __builtin_amdgcn_wmma_f32_16x16x32_bf16(false, a, false, b0, (short)0, acc0, false, false);
    acc1 = __builtin_amdgcn_wmma_f32_16x16x32_bf16(false, a, false, b1, (short)0, acc1, false, false);
    acc2 = __builtin_amdgcn_wmma_f32_16x16x32_bf16(false, a, false, b2, (short)0, acc2, false, false);
    acc3 = __builtin_amdgcn_wmma_f32_16x16x32_bf16(false, a, false, b3, (short)0, acc3, false, false);
  }

  // Epilogue: bias + ReLU + BatchNorm(eval), write bf16 h.
  // D layout: value r of lane l = element (M = r + (l&16 ? 8:0), Ncol = l&15).
  const int m0 = (lane & 16) ? 8 : 0;
  v8f accs[4] = {acc0, acc1, acc2, acc3};
#pragma unroll
  for (int t = 0; t < 4; ++t) {
    const int j  = j0 + t * 16 + (lane & 15);
    const int ch = k * HID + j;
    const float bias  = bl5[ch];
    const float scale = rsqrtf(run_var[ch] + BN_EPS) * gamma[ch];
    const float shift = beta[ch] - run_mean[ch] * scale;
#pragma unroll
    for (int r = 0; r < 8; ++r) {
      const int n = row0 + m0 + r;
      float v = accs[t][r] + bias;
      v = v > 0.0f ? v : 0.0f;           // ReLU
      v = v * scale + shift;             // BN (eval)
      hb[(size_t)n * CAT + ch] = (__bf16)v;
    }
  }
}

// ---------------------------------------------------------------------------
// GEMM2: out[n][j] = [mean2 | h](n,:) @ wcat2(j,:)^T + bias(j)
//        j<256 -> mu, j>=256 -> logvar. grid = (N/16, 1), block = 256.
// K = 5120, staged through LDS in 5 chunks of 1024.
// ---------------------------------------------------------------------------
__global__ void __launch_bounds__(256)
gemm2_kernel(const __bf16* __restrict__ meanb,   // [N][2560] (mean2)
             const __bf16* __restrict__ hb,      // [N][2560]
             const __bf16* __restrict__ wcat2,   // [512][5120]
             const float* __restrict__ bmu,
             const float* __restrict__ blv,
             float* __restrict__ out,            // mu [N][256] ++ logvar [N][256]
             int nNodes) {
  __shared__ __bf16 sA[16 * SA_STRIDE];

  const int rowTile = blockIdx.x;
  const int tid     = threadIdx.x;
  const int wave    = tid >> 5;
  const int lane    = tid & 31;
  const int row0    = rowTile * 16;
  const int j0      = wave * 64;

  const __bf16* aRowLds = sA + (lane & 15) * SA_STRIDE;
  const __bf16* bPtr[4];
#pragma unroll
  for (int t = 0; t < 4; ++t)
    bPtr[t] = wcat2 + (size_t)(j0 + t * 16 + (lane & 15)) * K2;

  v8f acc0 = {}, acc1 = {}, acc2 = {}, acc3 = {};

  const int sr  = tid >> 4;            // staging row 0..15
  const int sc0 = (tid & 15) * 64;     // staging col segment
  const int sn  = row0 + sr;

  for (int ck = 0; ck < K2 / CK; ++ck) {
    __syncthreads();                   // protect sA before overwrite
    {
      const int g0 = ck * CK + sc0;    // global K offset (64-aligned; 2560 ok)
      const __bf16* src = (g0 < CAT)
          ? (meanb + (size_t)sn * CAT + g0)
          : (hb + (size_t)sn * CAT + (g0 - CAT));
      stage64(src, sA + sr * SA_STRIDE + sc0);
    }
    stage_wait();

    const __bf16* bBase0 = bPtr[0] + ck * CK;
    const __bf16* bBase1 = bPtr[1] + ck * CK;
    const __bf16* bBase2 = bPtr[2] + ck * CK;
    const __bf16* bBase3 = bPtr[3] + ck * CK;
    for (int kt = 0; kt < CK / 32; ++kt) {
      v16bf a = load_frag(aRowLds + kt * 32, lane);
      v16bf b0 = load_frag(bBase0 + kt * 32, lane);
      v16bf b1 = load_frag(bBase1 + kt * 32, lane);
      v16bf b2 = load_frag(bBase2 + kt * 32, lane);
      v16bf b3 = load_frag(bBase3 + kt * 32, lane);
      acc0 = __builtin_amdgcn_wmma_f32_16x16x32_bf16(false, a, false, b0, (short)0, acc0, false, false);
      acc1 = __builtin_amdgcn_wmma_f32_16x16x32_bf16(false, a, false, b1, (short)0, acc1, false, false);
      acc2 = __builtin_amdgcn_wmma_f32_16x16x32_bf16(false, a, false, b2, (short)0, acc2, false, false);
      acc3 = __builtin_amdgcn_wmma_f32_16x16x32_bf16(false, a, false, b3, (short)0, acc3, false, false);
    }
  }

  const int m0 = (lane & 16) ? 8 : 0;
  v8f accs[4] = {acc0, acc1, acc2, acc3};
#pragma unroll
  for (int t = 0; t < 4; ++t) {
    const int j = j0 + t * 16 + (lane & 15);
    const bool isMu = (j < OUT_CH);
    const int jj = j & (OUT_CH - 1);
    const float bias = isMu ? bmu[jj] : blv[jj];
    float* base = out + (isMu ? 0 : (size_t)nNodes * OUT_CH);
#pragma unroll
    for (int r = 0; r < 8; ++r) {
      const int n = row0 + m0 + r;
      base[(size_t)n * OUT_CH + jj] = accs[t][r] + bias;
    }
  }
}

// ---------------------------------------------------------------------------
// Host-side orchestration
// ---------------------------------------------------------------------------
extern "C" void kernel_launch(void* const* d_in, const int* in_sizes, int n_in,
                              void* d_out, int out_size, void* d_ws, size_t ws_size,
                              hipStream_t stream) {
  const float* x        = (const float*)d_in[0];
  const int*   eidx     = (const int*)d_in[1];   // [2][E]
  const int*   eattr    = (const int*)d_in[2];   // [E]
  const float* Wl5      = (const float*)d_in[3];
  const float* Wr5      = (const float*)d_in[4];
  const float* bl5      = (const float*)d_in[5];
  const float* Wmu_l    = (const float*)d_in[6];
  const float* Wmu_r    = (const float*)d_in[7];
  const float* bmu      = (const float*)d_in[8];
  const float* Wlv_l    = (const float*)d_in[9];
  const float* Wlv_r    = (const float*)d_in[10];
  const float* blv      = (const float*)d_in[11];
  const float* gamma    = (const float*)d_in[12];
  const float* beta     = (const float*)d_in[13];
  const float* run_mean = (const float*)d_in[14];
  const float* run_var  = (const float*)d_in[15];
  float* out = (float*)d_out;

  const int nNodes = in_sizes[0] / IN_CH;   // 20000 (multiple of 16)
  const int E      = in_sizes[2];           // 100000
  const int* src = eidx;
  const int* dst = eidx + E;

  // ---- workspace layout (256B aligned slices) ----
  char* ws = (char*)d_ws;
  size_t off = 0;
  auto alloc = [&](size_t bytes) -> char* {
    char* p = ws + off;
    off += (bytes + 255) & ~(size_t)255;
    return p;
  };
  __bf16* xb    = (__bf16*)alloc((size_t)nNodes * IN_CH * sizeof(__bf16));
  __bf16* meanb = (__bf16*)alloc((size_t)nNodes * CAT * sizeof(__bf16));
  __bf16* hb    = (__bf16*)alloc((size_t)nNodes * CAT * sizeof(__bf16));
  float*  acc   = (float*)alloc((size_t)nNodes * CAT * sizeof(float));
  float*  cnt   = (float*)alloc((size_t)nNodes * 5 * sizeof(float));
  float*  cnt2  = (float*)alloc((size_t)nNodes * sizeof(float));
  __bf16* wcat1 = (__bf16*)alloc((size_t)5 * HID * K1 * sizeof(__bf16));
  __bf16* wcat2 = (__bf16*)alloc((size_t)2 * OUT_CH * K2 * sizeof(__bf16));
  (void)ws_size; (void)n_in; (void)out_size;

  // ---- zero accumulators (graph-capture-safe async memsets) ----
  (void)hipMemsetAsync(acc,  0, (size_t)nNodes * CAT * sizeof(float), stream);
  (void)hipMemsetAsync(cnt,  0, (size_t)nNodes * 5 * sizeof(float), stream);
  (void)hipMemsetAsync(cnt2, 0, (size_t)nNodes * sizeof(float), stream);

  // ---- precision conversions ----
  {
    int tot = nNodes * IN_CH;
    convert_x_kernel<<<(tot + 255) / 256, 256, 0, stream>>>(x, xb, tot);
  }
  {
    int tot = 5 * HID * K1;
    build_wcat1_kernel<<<(tot + 255) / 256, 256, 0, stream>>>(Wl5, Wr5, wcat1);
  }
  {
    int tot = 2 * OUT_CH * K2;
    build_wcat2_kernel<<<(tot + 255) / 256, 256, 0, stream>>>(Wmu_l, Wmu_r, Wlv_l, Wlv_r, wcat2);
  }

  // ---- layer 1: masked segment mean ----
  scatter1_kernel<<<E, 128, 0, stream>>>(x, src, dst, eattr, acc, cnt, E);
  {
    int tot = nNodes * CAT;
    mean1_kernel<<<(tot + 255) / 256, 256, 0, stream>>>(acc, cnt, meanb, tot);
  }
  // acc reused for layer 2: re-zero (stream-ordered after mean1 consumed it)
  (void)hipMemsetAsync(acc, 0, (size_t)nNodes * CAT * sizeof(float), stream);

  // ---- layer 1 GEMM + ReLU + BatchNorm -> h (bf16) ----
  gemm1_kernel<<<dim3(nNodes / 16, 5), 256, 0, stream>>>(
      meanb, xb, wcat1, bl5, gamma, beta, run_mean, run_var, hb);

  // ---- layer 2: uniform segment mean over h ----
  scatter2_kernel<<<E, 256, 0, stream>>>(hb, src, dst, acc, cnt2, E);
  {
    int tot = nNodes * CAT;
    mean2_kernel<<<(tot + 255) / 256, 256, 0, stream>>>(acc, cnt2, meanb, tot);
  }

  // ---- layer 2 GEMM -> mu ++ logvar ----
  gemm2_kernel<<<dim3(nNodes / 16, 1), 256, 0, stream>>>(
      meanb, hb, wcat2, bmu, blv, out, nNodes);
}
